// GCN_19464791786077
// MI455X (gfx1250) — matrix operand order repaired and checked by reading, not compiled, not run on hardware
//
#include <hip/hip_runtime.h>
#include <hip/hip_bf16.h>
#include <math.h>

typedef __attribute__((ext_vector_type(16))) __bf16       v16bf;
typedef __attribute__((ext_vector_type(8)))  float        v8f;
typedef __attribute__((ext_vector_type(4)))  unsigned int u32x4;

#define KDIM  128   // input features for every layer
#define MSLAB 64    // rows of X staged per block
#define XSTR  136   // padded LDS row stride (bf16 elems) -> 272B, rotates banks, keeps 16B align

// ---------------- degree / normalization ----------------
__global__ __launch_bounds__(256) void k_init_deg(float* deg, int n) {
    int i = blockIdx.x * 256 + threadIdx.x;
    if (i < n) deg[i] = 1.0f;                       // self-loop
}
__global__ __launch_bounds__(256) void k_count_deg(const long long* __restrict__ dst,
                                                   float* deg, int e) {
    int i = blockIdx.x * 256 + threadIdx.x;
    if (i < e) unsafeAtomicAdd(&deg[(int)dst[i]], 1.0f);
}
__global__ __launch_bounds__(256) void k_rsqrt(float* d, int n) {
    int i = blockIdx.x * 256 + threadIdx.x;
    if (i < n) d[i] = rsqrtf(d[i]);
}

// ---------------- W -> W^T bf16 ----------------
__global__ __launch_bounds__(256) void k_wt(const float* __restrict__ W,
                                            __bf16* __restrict__ wt, int nout) {
    int i = blockIdx.x * 256 + threadIdx.x;
    if (i < nout * KDIM) {
        int n = i / KDIM, k = i % KDIM;             // wt[n][k] = W[k][n]
        wt[i] = (__bf16)W[k * nout + n];
    }
}

// ---------------- WMMA GEMM: H = (relu?)X @ W ----------------
template <int NOUT>
__global__ __launch_bounds__(256) void k_gemm(const float* __restrict__ X,
                                              const __bf16* __restrict__ WT,
                                              float* __restrict__ H,
                                              int nrows, int relu_in) {
    __shared__ __bf16 xs[MSLAB * XSTR];             // 17,408 B
    __shared__ __bf16 bs[NOUT * XSTR];              // 34,816 B (NOUT=128)

    const int tid = threadIdx.x;
    const int m0  = blockIdx.x * MSLAB;

    // stage X slab, fp32 -> bf16, fused ReLU of previous layer
    for (int i = tid; i < MSLAB * KDIM; i += 256) {
        int r = i >> 7, k = i & 127;
        float v = (m0 + r < nrows) ? X[(size_t)(m0 + r) * KDIM + k] : 0.0f;
        if (relu_in) v = fmaxf(v, 0.0f);
        xs[r * XSTR + k] = (__bf16)v;
    }
    // stage W^T (contiguous global -> padded LDS)
    for (int i = tid; i < NOUT * KDIM; i += 256) {
        int n = i >> 7, k = i & 127;
        bs[n * XSTR + k] = WT[i];
    }
    __syncthreads();

    const int  wave  = tid >> 5, lane = tid & 31;
    const int  mtile = wave & 3;                    // 4 x 16-row tiles in slab
    const int  nh    = wave >> 2;                   // 2 column halves
    const int  l16   = lane & 15;
    const bool hi    = (lane & 16) != 0;
    constexpr int NT = NOUT / 32;                   // 16-col tiles per wave

    v8f acc[NT] = {};

#pragma unroll
    for (int kt = 0; kt < KDIM / 32; ++kt) {
        // A fragment 16x32 bf16: lanes 0-15 -> K 0-7 & 16-23; lanes 16-31 -> K 8-15 & 24-31
        const int ar = mtile * 16 + l16;
        const int ak = kt * 32 + (hi ? 8 : 0);
        union { v16bf v; u32x4 u[2]; } af;
        af.u[0] = *(const u32x4*)&xs[ar * XSTR + ak];
        af.u[1] = *(const u32x4*)&xs[ar * XSTR + ak + 16];
#pragma unroll
        for (int t = 0; t < NT; ++t) {
            // B fragment 32x16: lane = N column; lanes 0-15 -> K 0-7 & 16-23; lanes 16-31 -> K 8-15 & 24-31
            const int bn = nh * (NOUT / 2) + t * 16 + l16;
            const int bk = kt * 32 + (hi ? 8 : 0);
            union { v16bf v; u32x4 u[2]; } bfv;
            bfv.u[0] = *(const u32x4*)&bs[bn * XSTR + bk];
            bfv.u[1] = *(const u32x4*)&bs[bn * XSTR + bk + 16];
            acc[t] = __builtin_amdgcn_wmma_f32_16x16x32_bf16(
                false, af.v, false, bfv.v, (short)0, acc[t], false, false);
        }
    }

    // C/D layout: VGPR v, lanes 0-15 -> M=v; lanes 16-31 -> M=v+8; N = lane&15
#pragma unroll
    for (int t = 0; t < NT; ++t) {
        const int col = nh * (NOUT / 2) + t * 16 + l16;
        const int rb  = m0 + mtile * 16 + (hi ? 8 : 0);
#pragma unroll
        for (int v = 0; v < 8; ++v) {
            int row = rb + v;
            if (row < nrows) H[(size_t)row * NOUT + col] = acc[t][v];
        }
    }
}

// ---------------- self-loop + bias: O = dis^2 * H + b ----------------
template <int F>
__global__ __launch_bounds__(256) void k_selfbias(const float* __restrict__ H,
                                                  float* __restrict__ O,
                                                  const float* __restrict__ dis,
                                                  const float* __restrict__ b, int n) {
    int i = blockIdx.x * 256 + threadIdx.x;
    if (i < n * F) {
        int r = i / F, c = i % F;
        float d = dis[r];
        O[i] = d * d * H[i] + b[c];
    }
}

// ---------------- edge aggregation: O[dst] += dis[src]*dis[dst]*H[src] ----------------
template <int F>
__global__ __launch_bounds__(256) void k_aggregate(const float* __restrict__ H,
                                                   float* __restrict__ O,
                                                   const float* __restrict__ dis,
                                                   const long long* __restrict__ src,
                                                   const long long* __restrict__ dst,
                                                   int e) {
    constexpr int TPE = F / 4;                      // lanes per edge (float4 each)
    long long gid = (long long)blockIdx.x * 256 + threadIdx.x;
    long long ei  = gid / TPE;
    if (ei >= e) return;
    int c = (int)(gid % TPE) * 4;
    int s = (int)src[ei], d = (int)dst[ei];
    float coef = dis[s] * dis[d];
    const float4 hv = *(const float4*)&H[(size_t)s * F + c];
    float* op = &O[(size_t)d * F + c];
    unsafeAtomicAdd(op + 0, coef * hv.x);
    unsafeAtomicAdd(op + 1, coef * hv.y);
    unsafeAtomicAdd(op + 2, coef * hv.z);
    unsafeAtomicAdd(op + 3, coef * hv.w);
}

// ---------------- log_softmax over 64 cols, one wave per row ----------------
__global__ __launch_bounds__(256) void k_logsoftmax(float* __restrict__ O, int n) {
    int row  = blockIdx.x * 8 + (threadIdx.x >> 5);
    int lane = threadIdx.x & 31;
    if (row >= n) return;
    float* p = &O[(size_t)row * 64];
    float v0 = p[lane], v1 = p[lane + 32];
    float m = fmaxf(v0, v1);
    for (int o = 16; o; o >>= 1) m = fmaxf(m, __shfl_xor(m, o, 32));
    float s = expf(v0 - m) + expf(v1 - m);
    for (int o = 16; o; o >>= 1) s += __shfl_xor(s, o, 32);
    float L = m + logf(s);
    p[lane]      = v0 - L;
    p[lane + 32] = v1 - L;
}

extern "C" void kernel_launch(void* const* d_in, const int* in_sizes, int n_in,
                              void* d_out, int out_size, void* d_ws, size_t ws_size,
                              hipStream_t stream) {
    const float*     x   = (const float*)d_in[0];
    const long long* ei  = (const long long*)d_in[1];   // int64 (2,E)
    const float*     W0  = (const float*)d_in[2];
    const float*     b0  = (const float*)d_in[3];
    const float*     W1  = (const float*)d_in[4];
    const float*     b1  = (const float*)d_in[5];
    const float*     W2  = (const float*)d_in[6];
    const float*     b2  = (const float*)d_in[7];
    float*           out = (float*)d_out;

    const int N = in_sizes[0] / 128;
    const int E = in_sizes[1] / 2;
    const long long* src = ei;
    const long long* dst = ei + E;

    // workspace layout
    char*   w   = (char*)d_ws;
    size_t  o   = 0;
    float*  dis = (float*)(w + o); o += (((size_t)N * 4) + 255) & ~(size_t)255;
    __bf16* wt  = (__bf16*)(w + o); o += ((size_t)(128 * 128 * 2) + 255) & ~(size_t)255;
    float*  H   = (float*)(w + o); o += (((size_t)N * 128 * 4) + 255) & ~(size_t)255;
    float*  A   = (float*)(w + o);

    auto cdiv = [](long long a, long long b) { return (int)((a + b - 1) / b); };

    // degrees -> dis = rsqrt(indeg + 1)
    k_init_deg <<<cdiv(N, 256), 256, 0, stream>>>(dis, N);
    k_count_deg<<<cdiv(E, 256), 256, 0, stream>>>(dst, dis, E);
    k_rsqrt    <<<cdiv(N, 256), 256, 0, stream>>>(dis, N);

    // layer 0: h = x@W0 ; out = D^-1/2 (A+I) D^-1/2 h + b0   (relu deferred)
    k_wt<<<cdiv(128 * 128, 256), 256, 0, stream>>>(W0, wt, 128);
    k_gemm<128><<<cdiv(N, MSLAB), 256, 0, stream>>>(x, wt, H, N, 0);
    k_selfbias<128><<<cdiv((long long)N * 128, 256), 256, 0, stream>>>(H, A, dis, b0, N);
    k_aggregate<128><<<cdiv((long long)E * 32, 256), 256, 0, stream>>>(H, A, dis, src, dst, E);

    // layer 1 (relu of layer-0 output fused into GEMM stage-in)
    k_wt<<<cdiv(128 * 128, 256), 256, 0, stream>>>(W1, wt, 128);
    k_gemm<128><<<cdiv(N, MSLAB), 256, 0, stream>>>(A, wt, H, N, 1);
    k_selfbias<128><<<cdiv((long long)N * 128, 256), 256, 0, stream>>>(H, A, dis, b1, N);
    k_aggregate<128><<<cdiv((long long)E * 32, 256), 256, 0, stream>>>(H, A, dis, src, dst, E);

    // layer 2 -> d_out, then log_softmax in place
    k_wt<<<cdiv(128 * 64, 256), 256, 0, stream>>>(W2, wt, 64);
    k_gemm<64><<<cdiv(N, MSLAB), 256, 0, stream>>>(A, wt, H, N, 1);
    k_selfbias<64><<<cdiv((long long)N * 64, 256), 256, 0, stream>>>(H, out, dis, b2, N);
    k_aggregate<64><<<cdiv((long long)E * 16, 256), 256, 0, stream>>>(H, out, dis, src, dst, E);
    k_logsoftmax<<<cdiv(N, 8), 256, 0, stream>>>(out, N);
}
